// LSTM_Model_39548058861887
// MI455X (gfx1250) — matrix operand order repaired and checked by reading, not compiled, not run on hardware
//
#include <hip/hip_runtime.h>
#include <hip/hip_bf16.h>
#include <math.h>

// ---------------------------------------------------------------------------
// CDNA5 (gfx1250) 2-layer LSTM: wave32 WMMA bf16 + async global->LDS prefetch
// ---------------------------------------------------------------------------

typedef __attribute__((ext_vector_type(16))) __bf16 v16bf;
typedef __attribute__((ext_vector_type(8)))  __bf16 v8bf;
typedef __attribute__((ext_vector_type(8)))  float  v8f;

#define B_   32
#define T_   200
#define E_   300
#define H_   512
#define G4_  2048   // 4*H
#define NWG_ 32     // persistent workgroups; each owns H_/NWG_ = 16 h-columns

// ---- workspace layout (bytes) ----
// Xp is stored PERMUTED: Xp[t][wg][b*64 + gate*16 + cc]  (8KB contiguous per
// (t,wg) slice -> perfect for async global->LDS prefetch in the rnn kernel)
static constexpr size_t XP_OFF  = 0;
static constexpr size_t XP_SZ   = (size_t)T_ * B_ * G4_ * 4;
static constexpr size_t XPK_OFF = XP_OFF + XP_SZ;                 // bf16 W0x fragments
static constexpr size_t XPK_SZ  = (size_t)10 * 128 * 512 * 2;
static constexpr size_t L0_OFF  = XPK_OFF + XPK_SZ;               // bf16 W0h fragments
static constexpr size_t L0_SZ   = (size_t)16 * 32 * 4 * 512 * 2;
static constexpr size_t L1_OFF  = L0_OFF + L0_SZ;                 // bf16 W1 fragments
static constexpr size_t L1_SZ   = (size_t)32 * 32 * 4 * 512 * 2;
static constexpr size_t H0_OFF  = L1_OFF + L1_SZ;                 // bf16 h0[2][32][512]
static constexpr size_t H0_SZ   = (size_t)2 * B_ * H_ * 2;
static constexpr size_t H1_OFF  = H0_OFF + H0_SZ;                 // bf16 h1[2][32][512]
static constexpr size_t H1_SZ   = (size_t)2 * B_ * H_ * 2;
static constexpr size_t BAR_OFF = H1_OFF + H1_SZ;                 // 2x u32 barrier state
static constexpr size_t ZERO_BYTES = H0_SZ + H1_SZ + 256;

__device__ __forceinline__ float sigm(float x) { return 1.0f / (1.0f + __expf(-x)); }
__device__ __forceinline__ v8f vzero8() { v8f r = {}; return r; }

// A-matrix 16x32 bf16 fragment: lane L: row=L&15, half=L>>4;
// elems j = K[half*16 + j] (contiguous, matching the B packing).
__device__ __forceinline__ v16bf load_afrag(const __bf16* __restrict__ mat, int ldk,
                                            int row, int kbase, int half) {
  const __bf16* p0 = mat + (size_t)row * ldk + kbase + half * 16;
  v8bf lo = *(const v8bf*)p0;
  v8bf hi = *(const v8bf*)(p0 + 8);
  return __builtin_shufflevector(lo, hi, 0, 1, 2, 3, 4, 5, 6, 7,
                                 8, 9, 10, 11, 12, 13, 14, 15);
}

__device__ __forceinline__ v16bf load_bfrag(const __bf16* __restrict__ frag, int lane) {
  return *(const v16bf*)(frag + (size_t)lane * 16);
}

__device__ __forceinline__ v8f wmma_bf16(v16bf a, v16bf b, v8f c) {
  return __builtin_amdgcn_wmma_f32_16x16x32_bf16(false, a, false, b, (short)0, c,
                                                 false, false);
}

// ---- CDNA5 async global->LDS (ASYNCcnt-tracked) ------------------------------
// Generic LDS pointers carry the wave-relative LDS byte offset in addr[31:0].
__device__ __forceinline__ void async_ld_b128(void* lds_dst, const void* gsrc) {
  unsigned int       ldsoff = (unsigned int)(uintptr_t)lds_dst;
  unsigned long long gaddr  = (unsigned long long)(uintptr_t)gsrc;
  asm volatile("global_load_async_to_lds_b128 %0, %1, off"
               :
               : "v"(ldsoff), "v"(gaddr)
               : "memory");
}
#define WAIT_ASYNCCNT(N) asm volatile("s_wait_asynccnt " #N ::: "memory")

// ---------------------------------------------------------------------------
// Kernel 1: repack W0 / W1 into WMMA B-fragment order (bf16).
// B 32x16 lane layout: lane L: col = L&15; elems j = K[(L>>4)*16 + j].
// ---------------------------------------------------------------------------
#define NXP_FRAG (10 * 128)
#define NL0_FRAG (16 * 32 * 4)
#define NL1_FRAG (32 * 32 * 4)

__global__ void __launch_bounds__(256) pack_weights(const float* __restrict__ W0,
                                                    const float* __restrict__ W1,
                                                    __bf16* __restrict__ xpack,
                                                    __bf16* __restrict__ l0pack,
                                                    __bf16* __restrict__ l1pack) {
  int gw   = blockIdx.x * 8 + (threadIdx.x >> 5);
  int lane = threadIdx.x & 31;
  int half = lane >> 4, idx = lane & 15;

  __bf16 vals[16];
  __bf16* dst;

  if (gw < NXP_FRAG) {                       // W0 rows [0,300), zero-padded to 320
    int kc = gw / 128, ntile = gw % 128;
    int n = ntile * 16 + idx;
#pragma unroll
    for (int j = 0; j < 16; ++j) {
      int k = kc * 32 + half * 16 + j;
      vals[j] = (__bf16)((k < E_) ? W0[(size_t)k * G4_ + n] : 0.0f);
    }
    dst = xpack + (size_t)gw * 512 + lane * 16;
  } else if (gw < NXP_FRAG + NL0_FRAG) {     // W0 rows [300,812): recurrent part
    int f = gw - NXP_FRAG;                   // f = (kc*32 + wg)*4 + gate
    int gate = f & 3, wg = (f >> 2) & 31, kc = f >> 7;
    int n = gate * H_ + wg * 16 + idx;
#pragma unroll
    for (int j = 0; j < 16; ++j) {
      int k = E_ + kc * 32 + half * 16 + j;
      vals[j] = (__bf16)W0[(size_t)k * G4_ + n];
    }
    dst = l0pack + (size_t)f * 512 + lane * 16;
  } else {                                   // W1 [1024][2048]
    int f = gw - NXP_FRAG - NL0_FRAG;
    int gate = f & 3, wg = (f >> 2) & 31, kc = f >> 7;
    int n = gate * H_ + wg * 16 + idx;
#pragma unroll
    for (int j = 0; j < 16; ++j) {
      int k = kc * 32 + half * 16 + j;
      vals[j] = (__bf16)W1[(size_t)k * G4_ + n];
    }
    dst = l1pack + (size_t)f * 512 + lane * 16;
  }

  v8bf lo, hi;
#pragma unroll
  for (int j = 0; j < 8; ++j) { lo[j] = vals[j]; hi[j] = vals[j + 8]; }
  *(v8bf*)dst = lo;
  *(v8bf*)(dst + 8) = hi;
}

// ---------------------------------------------------------------------------
// Kernel 2: Xp = emb[x] @ W0[:300,:] + b0, written in per-(t,wg) slice layout.
// grid (400, 16), block 256 (8 waves, one ntile each).
// ---------------------------------------------------------------------------
__global__ void __launch_bounds__(256) xproj_kernel(const int* __restrict__ x_in,
                                                    const float* __restrict__ emb,
                                                    const float* __restrict__ b0,
                                                    const __bf16* __restrict__ xpack,
                                                    float* __restrict__ Xp) {
  __shared__ __bf16 Atile[16 * 32];

  int tid  = threadIdx.x;
  int lane = tid & 31, w = tid >> 5;
  int half = lane >> 4, idx = lane & 15;
  int mbase = blockIdx.x * 16;
  int ntile = blockIdx.y * 8 + w;

  v8f acc = vzero8();

  for (int kc = 0; kc < 10; ++kc) {
#pragma unroll
    for (int e = tid; e < 512; e += 256) {
      int row = e >> 5, k = e & 31;
      int kg = kc * 32 + k;
      int m = mbase + row;
      int t = m >> 5, b = m & 31;
      int tok = x_in[b * T_ + t];
      float v = (kg < E_) ? emb[(size_t)tok * E_ + kg] : 0.0f;
      Atile[row * 32 + k] = (__bf16)v;
    }
    __syncthreads();

    v16bf a  = load_afrag(Atile, 32, idx, 0, half);
    v16bf bf = load_bfrag(xpack + ((size_t)kc * 128 + ntile) * 512, lane);
    acc = wmma_bf16(a, bf, acc);
    __syncthreads();
  }

  // n -> (gate, wg, cc); store into permuted layout Xp[t][wg][b*64+gate*16+cc]
  int n = ntile * 16 + idx;
  int g = n >> 9, wgc = (n >> 4) & 31, cc = n & 15;
  float bias = b0[n];
#pragma unroll
  for (int v = 0; v < 8; ++v) {
    int m = mbase + half * 8 + v;
    int t = m >> 5, b = m & 31;
    Xp[((size_t)t * NWG_ + wgc) * (B_ * 64) + b * 64 + g * 16 + cc] = acc[v] + bias;
  }
}

// ---------------------------------------------------------------------------
// Device-wide generation barrier (persistent grid of NWG_ workgroups)
// ---------------------------------------------------------------------------
__device__ __forceinline__ void grid_barrier(unsigned* bar, int nwg) {
  __syncthreads();
  if (threadIdx.x == 0) {
    __threadfence();
    unsigned* cnt = bar;
    unsigned* gen = bar + 1;
    unsigned g = __hip_atomic_load(gen, __ATOMIC_RELAXED, __HIP_MEMORY_SCOPE_AGENT);
    unsigned a = __hip_atomic_fetch_add(cnt, 1u, __ATOMIC_ACQ_REL, __HIP_MEMORY_SCOPE_AGENT);
    if (a == (unsigned)(nwg - 1)) {
      __hip_atomic_store(cnt, 0u, __ATOMIC_RELAXED, __HIP_MEMORY_SCOPE_AGENT);
      __hip_atomic_fetch_add(gen, 1u, __ATOMIC_RELEASE, __HIP_MEMORY_SCOPE_AGENT);
    } else {
      while (__hip_atomic_load(gen, __ATOMIC_ACQUIRE, __HIP_MEMORY_SCOPE_AGENT) == g)
        __builtin_amdgcn_s_sleep(1);
    }
  }
  __syncthreads();
}

// ---------------------------------------------------------------------------
// Kernel 3: persistent recurrent kernel. 32 WGs x 256 threads.
// Wave w -> (gate = w&3, khalf = w>>2): each wave computes one gate tile over
// half of K; two disjoint LDS partial buffers (no LDS atomics); gate math sums
// them with the async-prefetched Xp slice. h state ping-pongs in global (L2).
// ---------------------------------------------------------------------------
__global__ void __launch_bounds__(256) lstm_seq(const float* __restrict__ Xp,
                                                const __bf16* __restrict__ l0pack,
                                                const __bf16* __restrict__ l1pack,
                                                const float* __restrict__ b1,
                                                __bf16* __restrict__ h0buf,
                                                __bf16* __restrict__ h1buf,
                                                unsigned* __restrict__ bar) {
  __shared__ __align__(16) float zx[2][B_ * 64];  // async-prefetched Xp slices
  __shared__ float zp[2][B_ * 64];                // per-K-half WMMA partials
  __shared__ float c0[B_ * 16];
  __shared__ float c1[B_ * 16];
  __shared__ float b1s[64];

  int tid  = threadIdx.x;
  int lane = tid & 31, w = tid >> 5;
  int half = lane >> 4, idx = lane & 15;
  int g  = w & 3;        // gate owned by this wave
  int kh = w >> 2;       // K-half owned by this wave
  int wg = blockIdx.x;

  for (int e = tid; e < B_ * 16; e += 256) { c0[e] = 0.0f; c1[e] = 0.0f; }
  if (tid < 64) b1s[tid] = b1[(tid >> 4) * H_ + wg * 16 + (tid & 15)];
  __syncthreads();

  // prologue: async-prefetch Xp slice for t=0 into zx[0] (2 x b128 per thread)
  {
    const float* src = Xp + ((size_t)0 * NWG_ + wg) * (B_ * 64) + tid * 8;
#pragma unroll
    for (int j = 0; j < 2; ++j) async_ld_b128(&zx[0][tid * 8 + j * 4], src + j * 4);
  }

  for (int t = 0; t < T_; ++t) {
    int p = t & 1;
    const __bf16* h0r = h0buf + p * (B_ * H_);
    __bf16*       h0w = h0buf + (1 - p) * (B_ * H_);
    const __bf16* h1r = h1buf + p * (B_ * H_);
    __bf16*       h1w = h1buf + (1 - p) * (B_ * H_);

    // async-prefetch next timestep's Xp slice into the other buffer, then
    // wait for this step's slice (async loads complete in order per wave).
    {
      int tn = (t + 1 < T_) ? (t + 1) : t;
      const float* src = Xp + ((size_t)tn * NWG_ + wg) * (B_ * 64) + tid * 8;
#pragma unroll
      for (int j = 0; j < 2; ++j)
        async_ld_b128(&zx[p ^ 1][tid * 8 + j * 4], src + j * 4);
    }
    WAIT_ASYNCCNT(2);

    // ================= layer 0:  z = Xp[t] + h0_{t-1} @ W0h =================
    {
      v8f acc0 = vzero8(), acc1 = vzero8();
#pragma unroll
      for (int j = 0; j < 8; ++j) {              // K = 512, this wave: 8 kc
        int kc = kh * 8 + j;
        int kbase = kc * 32;
        v16bf A0 = load_afrag(h0r, H_, idx,      kbase, half);
        v16bf A1 = load_afrag(h0r, H_, 16 + idx, kbase, half);
        v16bf Bf = load_bfrag(l0pack + (((size_t)kc * 32 + wg) * 4 + g) * 512, lane);
        acc0 = wmma_bf16(A0, Bf, acc0);
        acc1 = wmma_bf16(A1, Bf, acc1);
      }
#pragma unroll
      for (int v = 0; v < 8; ++v) {
        zp[kh][(half * 8 + v) * 64 + g * 16 + idx]        = acc0[v];
        zp[kh][(16 + half * 8 + v) * 64 + g * 16 + idx]   = acc1[v];
      }
    }
    __syncthreads();

#pragma unroll
    for (int e = tid; e < B_ * 16; e += 256) {
      int b = e >> 4, cc = e & 15;
      int o = b * 64 + cc;
      float zi = zx[p][o]      + zp[0][o]      + zp[1][o];
      float zj = zx[p][o + 16] + zp[0][o + 16] + zp[1][o + 16];
      float zf = zx[p][o + 32] + zp[0][o + 32] + zp[1][o + 32];
      float zo = zx[p][o + 48] + zp[0][o + 48] + zp[1][o + 48];
      float cn = c0[e] * sigm(zf + 1.0f) + sigm(zi) * tanhf(zj);
      c0[e] = cn;
      h0w[(size_t)b * H_ + wg * 16 + cc] = (__bf16)(tanhf(cn) * sigm(zo));
    }
    grid_barrier(bar, NWG_);

    // ================= layer 1:  z = b1 + [h0_t ; h1_{t-1}] @ W1 ============
    {
      const __bf16* src = kh ? h1r : h0w;        // kh=0 -> h0_t, kh=1 -> h1_{t-1}
      v8f acc0 = vzero8(), acc1 = vzero8();
#pragma unroll
      for (int j = 0; j < 16; ++j) {             // K = 1024, this wave: 16 kc
        int kc = kh * 16 + j;
        int kbase = j * 32;
        v16bf A0 = load_afrag(src, H_, idx,      kbase, half);
        v16bf A1 = load_afrag(src, H_, 16 + idx, kbase, half);
        v16bf Bf = load_bfrag(l1pack + (((size_t)kc * 32 + wg) * 4 + g) * 512, lane);
        acc0 = wmma_bf16(A0, Bf, acc0);
        acc1 = wmma_bf16(A1, Bf, acc1);
      }
#pragma unroll
      for (int v = 0; v < 8; ++v) {
        zp[kh][(half * 8 + v) * 64 + g * 16 + idx]        = acc0[v];
        zp[kh][(16 + half * 8 + v) * 64 + g * 16 + idx]   = acc1[v];
      }
    }
    __syncthreads();

#pragma unroll
    for (int e = tid; e < B_ * 16; e += 256) {
      int b = e >> 4, cc = e & 15;
      int o = b * 64 + cc;
      float zi = b1s[cc]      + zp[0][o]      + zp[1][o];
      float zj = b1s[16 + cc] + zp[0][o + 16] + zp[1][o + 16];
      float zf = b1s[32 + cc] + zp[0][o + 32] + zp[1][o + 32];
      float zo = b1s[48 + cc] + zp[0][o + 48] + zp[1][o + 48];
      float cn = c1[e] * sigm(zf + 1.0f) + sigm(zi) * tanhf(zj);
      c1[e] = cn;
      h1w[(size_t)b * H_ + wg * 16 + cc] = (__bf16)(tanhf(cn) * sigm(zo));
    }
    grid_barrier(bar, NWG_);
  }
}

// ---------------------------------------------------------------------------
// Kernel 4: logits = relu(h1_last @ Wd + bd), [32x512]@[512x2]
// ---------------------------------------------------------------------------
__global__ void dense_out(const __bf16* __restrict__ h1,
                          const float* __restrict__ Wd,
                          const float* __restrict__ bd,
                          float* __restrict__ out) {
  int tid = threadIdx.x;
  if (tid >= B_ * 2) return;
  int b = tid >> 1, c = tid & 1;
  float s = bd[c];
  for (int k = 0; k < H_; ++k)
    s += (float)h1[(size_t)b * H_ + k] * Wd[k * 2 + c];
  out[tid] = fmaxf(s, 0.0f);
}

// ---------------------------------------------------------------------------
extern "C" void kernel_launch(void* const* d_in, const int* in_sizes, int n_in,
                              void* d_out, int out_size, void* d_ws, size_t ws_size,
                              hipStream_t stream) {
  const int*   x_in = (const int*)d_in[0];
  const float* emb  = (const float*)d_in[1];
  const float* W0   = (const float*)d_in[2];
  const float* b0   = (const float*)d_in[3];
  const float* W1   = (const float*)d_in[4];
  const float* b1   = (const float*)d_in[5];
  const float* Wd   = (const float*)d_in[6];
  const float* bd   = (const float*)d_in[7];

  char* ws = (char*)d_ws;
  float*    Xp     = (float*)(ws + XP_OFF);
  __bf16*   xpack  = (__bf16*)(ws + XPK_OFF);
  __bf16*   l0pack = (__bf16*)(ws + L0_OFF);
  __bf16*   l1pack = (__bf16*)(ws + L1_OFF);
  __bf16*   h0buf  = (__bf16*)(ws + H0_OFF);
  __bf16*   h1buf  = (__bf16*)(ws + H1_OFF);
  unsigned* bar    = (unsigned*)(ws + BAR_OFF);

  // zero h-state ping-pong buffers + barrier state (graph-capture safe)
  hipMemsetAsync(ws + H0_OFF, 0, ZERO_BYTES, stream);

  // one-time weight repack into WMMA B-fragment order (bf16)
  int nfrag = NXP_FRAG + NL0_FRAG + NL1_FRAG;          // 7424 fragments
  pack_weights<<<(nfrag + 7) / 8, 256, 0, stream>>>(W0, W1, xpack, l0pack, l1pack);

  // parallel input projection: [6400 x 300] @ [300 x 2048] + b0 (permuted out)
  xproj_kernel<<<dim3((B_ * T_) / 16, G4_ / 128), 256, 0, stream>>>(x_in, emb, b0,
                                                                    xpack, Xp);

  // persistent recurrent kernel: 200 steps x 2 layers, grid-barrier synced
  lstm_seq<<<NWG_, 256, 0, stream>>>(Xp, l0pack, l1pack, b1, h0buf, h1buf, bar);

  // final dense + relu; last h1 was written to buffer (T_ & 1) == 0
  dense_out<<<1, 64, 0, stream>>>(h1buf + (T_ & 1) * (B_ * H_), Wd, bd, (float*)d_out);
}
